// MultiHeadAttention_67276367724649
// MI455X (gfx1250) — compile-verified, matrix operand order checked by reading, not run on hardware
//
#include <hip/hip_runtime.h>
#include <hip/hip_bf16.h>

// ---------------------------------------------------------------------------
// MHA for gfx1250 (MI455X): f16 WMMA (16x16x32) with f32 accumulation.
// f16 intermediates in workspace, double-buffered LDS staging, async
// global->LDS copies (ASYNCcnt) for the straight-copy tiles.
//   B=4, S=2048, D_MODEL=1024, H=16, DEPTH=64
// ---------------------------------------------------------------------------

typedef __attribute__((ext_vector_type(16))) _Float16 v16h;
typedef __attribute__((ext_vector_type(8)))  _Float16 v8h;
typedef __attribute__((ext_vector_type(8)))  float    v8f;

#define SEQ    2048
#define DMODEL 1024
#define NHEAD  16
#define DEPTH  64
#define BATCH  4
#define MROWS  (BATCH * SEQ)   // 8192

__device__ __forceinline__ v8f wmma16(v16h a, v16h b, v8f c) {
  // D(16x16 f32) = A(16x32 f16) * B(32x16 f16) + C
  return __builtin_amdgcn_wmma_f32_16x16x32_f16(
      /*neg_a=*/false, a, /*neg_b=*/false, b,
      /*c_mod=*/(short)0, c, /*reuse_a=*/false, /*reuse_b=*/false);
}

// Per-lane 16-bit fragment: elements 0..7 from p0, 8..15 from p1
// (the two contiguous 8-element K-ranges of the ISA layout).
__device__ __forceinline__ v16h frag_ld(const _Float16* p0, const _Float16* p1) {
  v8h lo = *(const v8h*)p0;
  v8h hi = *(const v8h*)p1;
  return __builtin_shufflevector(lo, hi, 0, 1, 2, 3, 4, 5, 6, 7,
                                 8, 9, 10, 11, 12, 13, 14, 15);
}

// Convert float4 -> 4 f16 and store as one 8-byte LDS write.
__device__ __forceinline__ void st4h(_Float16* dst, float4 v) {
  union { _Float16 h[4]; unsigned long long u; } pk;
  pk.h[0] = (_Float16)v.x; pk.h[1] = (_Float16)v.y;
  pk.h[2] = (_Float16)v.z; pk.h[3] = (_Float16)v.w;
  *(unsigned long long*)dst = pk.u;
}

// Async 16B global -> LDS copy (GLOBAL_LOAD_ASYNC_TO_LDS_B128, ASYNCcnt).
// Generic pointers to __shared__ carry the wave-relative LDS offset in the
// low 32 bits (ISA 10.2 aperture mapping), which is what VDST expects.
__device__ __forceinline__ void async_cp16(void* lds_dst, const void* gsrc) {
  unsigned ldsa = (unsigned)(unsigned long long)lds_dst;
  asm volatile("global_load_async_to_lds_b128 %0, %1, off"
               :: "v"(ldsa), "v"(gsrc) : "memory");
}

__device__ __forceinline__ void async_wait0() {
  asm volatile("s_wait_asynccnt 0" ::: "memory");
}

// ---------------------------------------------------------------------------
// GEMM + bias:  C[M,1024] = A[M,1024] x W[1024,1024] + bias   (f16 math)
// TIN in {float, _Float16}; TOUT in {float, _Float16}; epilogue scaled.
// Block: 128 threads (4 waves). Block tile 128(M) x 64(N), K chunks of 64,
// double-buffered LDS. Wave w computes rows [m0 + 32w, +32) over 64 columns.
// A staged row-major f16 (async copy when input is already f16);
// W staged TRANSPOSED f16 ([n][k]) so fragments are contiguous b128 reads.
// ---------------------------------------------------------------------------
template <typename TIN, typename TOUT>
__global__ __launch_bounds__(128)
void mha_gemm_bias(const TIN* __restrict__ A, const float* __restrict__ W,
                   const float* __restrict__ bias, TOUT* __restrict__ C,
                   float oscale) {
  __shared__ _Float16 Alds[2][128][72];   // [buf][m][k], +8 pad
  __shared__ _Float16 Wt[2][64][72];      // [buf][n][k]

  const int wave = threadIdx.x >> 5;
  const int lane = threadIdx.x & 31;
  const int half = lane >> 4;
  const int lrow = lane & 15;

  const int m0 = blockIdx.y * 128;
  const int n0 = blockIdx.x * 64;

  auto stage = [&](int kt, int buf) {
    const TIN*   Ag = A + (size_t)m0 * DMODEL + kt * 64;
    const float* Wg = W + (size_t)(kt * 64) * DMODEL + n0;
    if constexpr (sizeof(TIN) == 4) {     // f32 input: convert while staging
#pragma unroll
      for (int i = 0; i < 16; ++i) {      // 128x64 floats = 2048 float4s
        int f = threadIdx.x + i * 128;
        int row = f >> 4, c4 = (f & 15) * 4;
        st4h(&Alds[buf][row][c4],
             *(const float4*)((const float*)Ag + (size_t)row * DMODEL + c4));
      }
    } else {                              // f16 input: async b128 copies
#pragma unroll
      for (int i = 0; i < 8; ++i) {       // 128x64 halves = 1024 x 16B
        int f = threadIdx.x + i * 128;
        int row = f >> 3, c8 = (f & 7) * 8;
        async_cp16(&Alds[buf][row][c8],
                   (const _Float16*)Ag + (size_t)row * DMODEL + c8);
      }
    }
#pragma unroll
    for (int i = 0; i < 8; ++i) {         // W tile 64x64, transposed store
      int f = threadIdx.x + i * 128;
      int row = f >> 4, c4 = (f & 15) * 4;   // row = k, c4.. = n
      float4 w4 = *(const float4*)(Wg + (size_t)row * DMODEL + c4);
      Wt[buf][c4 + 0][row] = (_Float16)w4.x;
      Wt[buf][c4 + 1][row] = (_Float16)w4.y;
      Wt[buf][c4 + 2][row] = (_Float16)w4.z;
      Wt[buf][c4 + 3][row] = (_Float16)w4.w;
    }
  };

  v8f acc[2][4] = {};   // 2 M-subtiles x 4 N-subtiles

  stage(0, 0);
  for (int kt = 0; kt < DMODEL / 64; ++kt) {
    async_wait0();       // our async copies into the upcoming buffer are done
    __syncthreads();     // publish block-wide (also waits DS stores)
    const int cur = kt & 1;
    if (kt + 1 < DMODEL / 64) stage(kt + 1, cur ^ 1);

#pragma unroll
    for (int t = 0; t < 2; ++t) {         // 2 K-steps of 32
      const int k0 = 32 * t + 8 * half;
      v16h a0 = frag_ld(&Alds[cur][wave * 32 + lrow][k0],
                        &Alds[cur][wave * 32 + lrow][k0 + 16]);
      v16h a1 = frag_ld(&Alds[cur][wave * 32 + 16 + lrow][k0],
                        &Alds[cur][wave * 32 + 16 + lrow][k0 + 16]);
#pragma unroll
      for (int sub = 0; sub < 4; ++sub) {
        v16h bf = frag_ld(&Wt[cur][sub * 16 + lrow][k0],
                          &Wt[cur][sub * 16 + lrow][k0 + 16]);
        acc[0][sub] = wmma16(a0, bf, acc[0][sub]);
        acc[1][sub] = wmma16(a1, bf, acc[1][sub]);
      }
    }
  }

#pragma unroll
  for (int ms = 0; ms < 2; ++ms)
#pragma unroll
    for (int sub = 0; sub < 4; ++sub) {
      float bv = bias[n0 + sub * 16 + lrow];
#pragma unroll
      for (int r = 0; r < 8; ++r) {
        int row = m0 + wave * 32 + ms * 16 + r + 8 * half;
        C[(size_t)row * DMODEL + n0 + sub * 16 + lrow] =
            (TOUT)((acc[ms][sub][r] + bv) * oscale);
      }
    }
}

// ---------------------------------------------------------------------------
// Flash attention over f16 Q/K/V (1/sqrt(depth) pre-folded into Q).
// Block = (query-tile of 64, head, batch), 128 threads; wave w owns rows
// [qt*64+16w, +16). K staged [key][depth] via ASYNC copies, V staged
// TRANSPOSED [depth][key] via register path; both double-buffered.
// Online softmax in f32; P round-trips through a per-wave f16 LDS strip to
// become A-layout for P*V. Output stored f16.
// ---------------------------------------------------------------------------
__global__ __launch_bounds__(128)
void mha_flash_attn_f16(const _Float16* __restrict__ Qp,
                        const _Float16* __restrict__ Kp,
                        const _Float16* __restrict__ Vp,
                        _Float16* __restrict__ Ao) {
  __shared__ _Float16 Klds[2][64][72];     // [buf][key][depth]
  __shared__ _Float16 Vt[2][64][72];       // [buf][depth][key]
  __shared__ _Float16 Plds[4][16][72];     // per-wave [qrow][key]

  const int qt = blockIdx.x;
  const int h  = blockIdx.y;
  const int b  = blockIdx.z;
  const int wave = threadIdx.x >> 5;
  const int lane = threadIdx.x & 31;
  const int half = lane >> 4;
  const int lrow = lane & 15;

  const int s0 = qt * 64 + wave * 16;

  // Q fragments: direct f16 global loads (scale already folded in).
  v16h qfrag[2];
  {
    const _Float16* Qb = Qp + ((size_t)b * SEQ + s0 + lrow) * DMODEL + h * DEPTH;
#pragma unroll
    for (int t = 0; t < 2; ++t) {
      const int base = 32 * t + 8 * half;
      qfrag[t] = frag_ld(Qb + base, Qb + base + 16);
    }
  }

  auto stage = [&](int kt, int buf) {
    const _Float16* Kg = Kp + ((size_t)b * SEQ + kt * 64) * DMODEL + h * DEPTH;
    const _Float16* Vg = Vp + ((size_t)b * SEQ + kt * 64) * DMODEL + h * DEPTH;
#pragma unroll
    for (int i = 0; i < 4; ++i) {          // 64x64 halves = 512 x 16B
      int f = threadIdx.x + i * 128;
      int row = f >> 3, c8 = (f & 7) * 8;  // row = key, c8.. = depth
      async_cp16(&Klds[buf][row][c8], Kg + (size_t)row * DMODEL + c8);
      v8h vv = *(const v8h*)(Vg + (size_t)row * DMODEL + c8);
#pragma unroll
      for (int j = 0; j < 8; ++j) Vt[buf][c8 + j][row] = vv[j];
    }
  };

  v8f o[4] = {};                           // O strip 16x64 (C layout, f32)
  float m_[8], l_[8];
#pragma unroll
  for (int r = 0; r < 8; ++r) { m_[r] = -INFINITY; l_[r] = 0.0f; }

  stage(0, 0);
  for (int kt = 0; kt < SEQ / 64; ++kt) {
    async_wait0();       // our K-tile async copies are complete
    __syncthreads();     // publish block-wide (also waits V DS stores)
    const int cur = kt & 1;
    if (kt + 1 < SEQ / 64) stage(kt + 1, cur ^ 1);

    // ---- S = Q * K^T : B[k=depth][n=key] -> read Klds[key][depth] ----
    v8f s[4] = {};
#pragma unroll
    for (int t = 0; t < 2; ++t) {
      const int k0 = 32 * t + 8 * half;
#pragma unroll
      for (int sub = 0; sub < 4; ++sub) {
        v16h bf = frag_ld(&Klds[cur][sub * 16 + lrow][k0],
                          &Klds[cur][sub * 16 + lrow][k0 + 16]);
        s[sub] = wmma16(qfrag[t], bf, s[sub]);
      }
    }

    // ---- online softmax (rows r + 8*half; 16-lane-half reductions) ----
#pragma unroll
    for (int r = 0; r < 8; ++r) {
      float tm = fmaxf(fmaxf(s[0][r], s[1][r]), fmaxf(s[2][r], s[3][r]));
      tm = fmaxf(tm, __shfl_xor(tm, 1));
      tm = fmaxf(tm, __shfl_xor(tm, 2));
      tm = fmaxf(tm, __shfl_xor(tm, 4));
      tm = fmaxf(tm, __shfl_xor(tm, 8));
      float mnew  = fmaxf(m_[r], tm);
      float alpha = __expf(m_[r] - mnew);
      float rs = 0.0f;
#pragma unroll
      for (int sub = 0; sub < 4; ++sub) {
        float p = __expf(s[sub][r] - mnew);
        s[sub][r] = p;
        rs += p;
      }
      rs += __shfl_xor(rs, 1);
      rs += __shfl_xor(rs, 2);
      rs += __shfl_xor(rs, 4);
      rs += __shfl_xor(rs, 8);
      l_[r] = l_[r] * alpha + rs;
      m_[r] = mnew;
#pragma unroll
      for (int sub = 0; sub < 4; ++sub) o[sub][r] *= alpha;
    }

    // ---- P (C-layout f32) -> per-wave LDS f16 strip [qrow][key] ----
#pragma unroll
    for (int sub = 0; sub < 4; ++sub)
#pragma unroll
      for (int r = 0; r < 8; ++r)
        Plds[wave][r + 8 * half][sub * 16 + lrow] = (_Float16)s[sub][r];

    // ---- O += P * V : B[k=key][n=depth] -> read Vt[depth][key] ----
#pragma unroll
    for (int t = 0; t < 2; ++t) {
      const int k0 = 32 * t + 8 * half;
      v16h pa = frag_ld(&Plds[wave][lrow][k0],
                        &Plds[wave][lrow][k0 + 16]);
#pragma unroll
      for (int sub = 0; sub < 4; ++sub) {
        v16h vf = frag_ld(&Vt[cur][sub * 16 + lrow][k0],
                          &Vt[cur][sub * 16 + lrow][k0 + 16]);
        o[sub] = wmma16(pa, vf, o[sub]);
      }
    }
  }

  // ---- normalize and write f16 concat layout [B,S,1024] ----
  float inv[8];
#pragma unroll
  for (int r = 0; r < 8; ++r) inv[r] = 1.0f / l_[r];
  _Float16* Ob = Ao + ((size_t)b * SEQ + s0) * DMODEL + h * DEPTH;
#pragma unroll
  for (int sub = 0; sub < 4; ++sub)
#pragma unroll
    for (int r = 0; r < 8; ++r)
      Ob[(size_t)(r + 8 * half) * DMODEL + sub * 16 + lrow] =
          (_Float16)(o[sub][r] * inv[r]);
}

// ---------------------------------------------------------------------------
// Host-side launch
// in order: v, k, q, wq_w, wq_b, wk_w, wk_b, wv_w, wv_b, wo_w, wo_b
// ---------------------------------------------------------------------------
extern "C" void kernel_launch(void* const* d_in, const int* in_sizes, int n_in,
                              void* d_out, int out_size, void* d_ws, size_t ws_size,
                              hipStream_t stream) {
  (void)in_sizes; (void)n_in; (void)out_size; (void)ws_size;

  const float* v    = (const float*)d_in[0];
  const float* k    = (const float*)d_in[1];
  const float* q    = (const float*)d_in[2];
  const float* wq_w = (const float*)d_in[3];
  const float* wq_b = (const float*)d_in[4];
  const float* wk_w = (const float*)d_in[5];
  const float* wk_b = (const float*)d_in[6];
  const float* wv_w = (const float*)d_in[7];
  const float* wv_b = (const float*)d_in[8];
  const float* wo_w = (const float*)d_in[9];
  const float* wo_b = (const float*)d_in[10];

  const size_t NE = (size_t)MROWS * DMODEL;   // 8M elements per tensor
  _Float16* Qp   = (_Float16*)d_ws;
  _Float16* Kp   = Qp + NE;
  _Float16* Vp   = Kp + NE;
  _Float16* Aatt = Vp + NE;
  float* out = (float*)d_out;

  dim3 gemm_grid(DMODEL / 64, MROWS / 128);   // (16, 64)
  dim3 gemm_block(128);

  // 1/sqrt(DEPTH) folded into the Q projection epilogue (exact for softmax).
  mha_gemm_bias<float, _Float16><<<gemm_grid, gemm_block, 0, stream>>>(
      q, wq_w, wq_b, Qp, 0.125f);
  mha_gemm_bias<float, _Float16><<<gemm_grid, gemm_block, 0, stream>>>(
      k, wk_w, wk_b, Kp, 1.0f);
  mha_gemm_bias<float, _Float16><<<gemm_grid, gemm_block, 0, stream>>>(
      v, wv_w, wv_b, Vp, 1.0f);

  dim3 attn_grid(SEQ / 64, NHEAD, BATCH);     // (32, 16, 4)
  mha_flash_attn_f16<<<attn_grid, dim3(128), 0, stream>>>(Qp, Kp, Vp, Aatt);

  mha_gemm_bias<_Float16, float><<<gemm_grid, gemm_block, 0, stream>>>(
      Aatt, wo_w, wo_b, out, 1.0f);
}
